// Inference_18107582119992
// MI455X (gfx1250) — matrix operand order repaired and checked by reading, not compiled, not run on hardware
//
#include <hip/hip_runtime.h>

// ConvLSTM cell, fully fused. Per batch b: gate_g = W_g[128x128] * X_b[128x4096]
// (K = C channels, N = HW pixels), bf16 WMMA + f32 accumulation, fused
// sigmoid/tanh/state-update epilogue.
//
// Kernel 1 packs all 4 gate weight matrices f32->bf16 into d_ws (128 KB, done
// once per launch). Kernel 2 keeps all 4 bf16 weight matrices LDS-resident
// (128 KB of the 320 KB/WGP), staged via async global->LDS (ASYNCcnt), reads X
// once, writes hidden+state once. 512 threads (16 waves), 64 pixels per block.

#define BATCH 32
#define CIN   128
#define HIDN  128
#define HWPX  4096
#define NT    64      // pixels per block tile
#define XPAD  132     // padded row stride (bf16 elems) for transposed X tile

typedef __attribute__((ext_vector_type(16))) __bf16 bf16x16;
typedef __attribute__((ext_vector_type(8)))  float  v8f;
typedef __attribute__((ext_vector_type(4)))  int    v4i;

union Frag { int i[8]; bf16x16 v; };

#if __has_builtin(__builtin_amdgcn_global_load_async_to_lds_b128)
#define ASYNC_LDS 1
#else
#define ASYNC_LDS 0
#endif

// Pack two f32 into two bf16 (truncation) with a single v_perm_b32.
__device__ __forceinline__ unsigned pack_bf16(float lo, float hi) {
  return __builtin_amdgcn_perm(__float_as_uint(hi), __float_as_uint(lo),
                               0x07060302u);
}

__device__ __forceinline__ float sigmoidf_(float x) {
  return 1.0f / (1.0f + __expf(-x));
}

__device__ __forceinline__ float fast_tanh(float x) {
#if __has_builtin(__builtin_amdgcn_tanhf)
  return __builtin_amdgcn_tanhf(x);       // v_tanh_f32 (CDNA5 transcendental)
#else
  float e = __expf(-2.0f * x);
  return (1.0f - e) / (1.0f + e);
#endif
}

// ---------------------------------------------------------------------------
// Prep: convert all 4 gate weight matrices to bf16 in workspace (gate-major).
__global__ __launch_bounds__(256)
void pack_weights_kernel(const float* __restrict__ w_f, const float* __restrict__ w_i,
                         const float* __restrict__ w_n, const float* __restrict__ w_o,
                         unsigned* __restrict__ wbf)
{
  int g = blockIdx.x >> 4;                       // 16 blocks per gate
  const float* w = (g == 0) ? w_f : (g == 1) ? w_i : (g == 2) ? w_n : w_o;
  int i4 = ((blockIdx.x & 15) << 8) + threadIdx.x;   // float4 idx in gate, 0..4095
  float4 v = ((const float4*)w)[i4];
  uint2 p;
  p.x = pack_bf16(v.x, v.y);
  p.y = pack_bf16(v.z, v.w);
  ((uint2*)wbf)[(g << 12) + i4] = p;             // 4096 uint2 per gate
}

// ---------------------------------------------------------------------------
__global__ __launch_bounds__(512)
void convlstm_cell_kernel(const float* __restrict__ xin,
                          const float* __restrict__ state,
                          const unsigned* __restrict__ wbf,
                          const float* __restrict__ b_f, const float* __restrict__ b_i,
                          const float* __restrict__ b_n, const float* __restrict__ b_o,
                          float* __restrict__ out_h, float* __restrict__ out_s)
{
  __shared__ alignas(16) unsigned short sW[4 * HIDN * CIN];   // 128 KB, all 4 gates
  __shared__ alignas(16) unsigned short sXT[NT * XPAD];       // x tile [pix][c]
  __shared__ float sBias[4 * HIDN];

  const int tid  = threadIdx.x;
  const int lane = tid & 31;
  const int wv   = tid >> 5;            // wave 0..15
  const int mg   = wv & 7;              // M group (16 output channels)
  const int ng   = wv >> 3;             // N group (32 pixels)
  const int m0   = mg * 16;
  const int nb   = ng * 32;
  const int bb   = blockIdx.x >> 6;     // batch
  const int pix0 = (blockIdx.x & 63) * NT;

  // --- stage biases -------------------------------------------------------
  if (tid < HIDN) {
    sBias[0 * HIDN + tid] = b_f[tid];
    sBias[1 * HIDN + tid] = b_i[tid];
    sBias[2 * HIDN + tid] = b_n[tid];
    sBias[3 * HIDN + tid] = b_o[tid];
  }

  // --- stage X tile transposed into LDS: pairs along C -> one v_perm each --
  #pragma unroll
  for (int i = 0; i < 8; ++i) {
    int idx = tid + i * 512;            // 0..4095 (c-pair, pixel)
    int cp  = idx >> 6;                 // channel pair 0..63
    int n   = idx & 63;                 // pixel within tile (coalesced)
    const float* xp = xin + ((size_t)bb * CIN + cp * 2) * HWPX + pix0 + n;
    float a = xp[0];
    float b = xp[HWPX];
    *(unsigned*)&sXT[n * XPAD + cp * 2] = pack_bf16(a, b);
  }

  // --- stage all 4 gate weight matrices (async global->LDS, bf16) ---------
#if ASYNC_LDS
  #pragma unroll
  for (int i = 0; i < 16; ++i) {
    int o16 = tid + i * 512;            // uint4 index, 8192 * 16B = 128 KB
    const uint4* gp = (const uint4*)wbf + o16;
    uint4*       lp = (uint4*)sW + o16;
    __builtin_amdgcn_global_load_async_to_lds_b128(
        (__attribute__((address_space(1))) v4i*)gp,
        (__attribute__((address_space(3))) v4i*)lp, 0, 0);
  }
#if __has_builtin(__builtin_amdgcn_s_wait_asynccnt)
  __builtin_amdgcn_s_wait_asynccnt(0);
#else
  asm volatile("s_wait_asynccnt 0" ::: "memory");
#endif
#else
  #pragma unroll
  for (int i = 0; i < 16; ++i) {
    int o16 = tid + i * 512;
    ((uint4*)sW)[o16] = ((const uint4*)wbf)[o16];
  }
#endif

  const int nidx = lane & 15;
  const int hf   = lane >> 4;

  // --- hoist state loads (independent of LDS) to overlap the barrier -------
  float st[2][8];
  #pragma unroll
  for (int t = 0; t < 2; ++t) {
    #pragma unroll
    for (int r = 0; r < 8; ++r) {
      size_t idx = ((size_t)bb * HIDN + (m0 + r + 8 * hf)) * HWPX +
                   (pix0 + nb + t * 16 + nidx);
      st[t][r] = state[idx];
    }
  }

  __syncthreads();

  // --- B fragments: shared by all 4 gates, loaded once ---------------------
  Frag bfr[4][2];                       // [k-step][n-tile]
  #pragma unroll
  for (int kk = 0; kk < 4; ++kk) {
    #pragma unroll
    for (int t = 0; t < 2; ++t) {
      #pragma unroll
      for (int v = 0; v < 8; ++v) {
        int k0 = kk * 32 + (v & 3) * 2 + hf * 8 + (v >> 2) * 16;  // same k order as A
        bfr[kk][t].i[v] =
            *(const int*)(sXT + (nb + t * 16 + nidx) * XPAD + k0);
      }
    }
  }

  v8f acc[4][2];
  {
    v8f z = {};
    #pragma unroll
    for (int g = 0; g < 4; ++g) { acc[g][0] = z; acc[g][1] = z; }
  }

  // --- 32 WMMAs: 4 gates x 4 K-steps x 2 N-tiles ---------------------------
  #pragma unroll
  for (int g = 0; g < 4; ++g) {
    #pragma unroll
    for (int kk = 0; kk < 4; ++kk) {
      Frag af;
      #pragma unroll
      for (int v = 0; v < 8; ++v) {
        int k0 = kk * 32 + (v & 3) * 2 + hf * 8 + (v >> 2) * 16;  // ISA A layout
        af.i[v] = *(const int*)(sW + ((g * HIDN) + m0 + nidx) * CIN + k0);
      }
      acc[g][0] = __builtin_amdgcn_wmma_f32_16x16x32_bf16(
          false, af.v, false, bfr[kk][0].v, (short)0, acc[g][0], false, false);
      acc[g][1] = __builtin_amdgcn_wmma_f32_16x16x32_bf16(
          false, af.v, false, bfr[kk][1].v, (short)0, acc[g][1], false, false);
    }
  }

  // --- per-wave bias slice: 2x ds_load_b128 per gate -----------------------
  float biasr[4][8];
  #pragma unroll
  for (int g = 0; g < 4; ++g) {
    const float* bp = &sBias[g * HIDN + m0 + 8 * hf];
    *(float4*)&biasr[g][0] = *(const float4*)bp;
    *(float4*)&biasr[g][4] = *(const float4*)(bp + 4);
  }

  // --- fused epilogue: gates + state update --------------------------------
  // D layout: VGPR r -> M = r + 8*hf, N = lane&15
  #pragma unroll
  for (int t = 0; t < 2; ++t) {
    #pragma unroll
    for (int r = 0; r < 8; ++r) {
      int o   = m0 + r + 8 * hf;
      int pix = pix0 + nb + t * 16 + nidx;
      size_t idx = ((size_t)bb * HIDN + o) * HWPX + pix;

      float vf = acc[0][t][r] + biasr[0][r];
      float vi = acc[1][t][r] + biasr[1][r];
      float vn = acc[2][t][r] + biasr[2][r];
      float vo = acc[3][t][r] + biasr[3][r];

      float fg = sigmoidf_(vf);
      float ig = sigmoidf_(vi);
      float nw = fast_tanh(vn);
      float og = sigmoidf_(vo);

      float ns = st[t][r] * fg + ig * nw;
      out_s[idx] = ns;
      out_h[idx] = fast_tanh(ns) * og;
    }
  }
}

// ---------------------------------------------------------------------------
extern "C" void kernel_launch(void* const* d_in, const int* in_sizes, int n_in,
                              void* d_out, int out_size, void* d_ws, size_t ws_size,
                              hipStream_t stream) {
  (void)in_sizes; (void)n_in; (void)out_size; (void)ws_size;
  const float* xin   = (const float*)d_in[0];
  // d_in[1] = hidden (unused by the reference forward)
  const float* state = (const float*)d_in[2];
  const float* w_f   = (const float*)d_in[3];
  const float* b_f   = (const float*)d_in[4];
  const float* w_i   = (const float*)d_in[5];
  const float* b_i   = (const float*)d_in[6];
  const float* w_n   = (const float*)d_in[7];
  const float* b_n   = (const float*)d_in[8];
  const float* w_o   = (const float*)d_in[9];
  const float* b_o   = (const float*)d_in[10];

  float* out_h = (float*)d_out;                        // new_hidden first
  float* out_s = out_h + (size_t)BATCH * HIDN * HWPX;  // then new_state

  // d_ws usage: 128 KB for packed bf16 weights (harness scratch covers this).
  unsigned* wbf = (unsigned*)d_ws;
  pack_weights_kernel<<<dim3(64), dim3(256), 0, stream>>>(w_f, w_i, w_n, w_o, wbf);

  dim3 grid(BATCH * (HWPX / NT));                      // 32 * 64 = 2048
  dim3 block(512);                                     // 16 waves
  convlstm_cell_kernel<<<grid, block, 0, stream>>>(
      xin, state, wbf, b_f, b_i, b_n, b_o, out_h, out_s);
}
